// HeteroRelConv_64484638982806
// MI455X (gfx1250) — matrix operand order repaired and checked by reading, not compiled, not run on hardware
//
#include <hip/hip_runtime.h>
#include <hip/hip_bf16.h>
#include <hip/hip_fp16.h>

// ---------------------------------------------------------------------------
// Problem constants (match reference)
// ---------------------------------------------------------------------------
static constexpr int CNA  = 200000;
static constexpr int CNB  = 300000;
static constexpr int CNM  = 100000;
static constexpr int CE   = 500000;
static constexpr int CD   = 64;
static constexpr int CG   = 512;
static constexpr int CNL  = 3;
static constexpr int CNET = 8;

// edge types: 0=atom,1=bond,2=motif
static const int H_SRC_T[8] = {0, 1, 2, 2, 2, 1, 1, 0};
static const int H_DST_T[8] = {2, 2, 2, 1, 0, 1, 0, 0};
static const int H_NSZ[3]   = {CNA, CNB, CNM};

typedef __attribute__((ext_vector_type(16))) _Float16 v16h;
typedef __attribute__((ext_vector_type(8)))  float    v8f;

// ---------------------------------------------------------------------------
// Weight convert + transpose: W[in][out] f32 -> WT[out][in] f16
// ---------------------------------------------------------------------------
__global__ __launch_bounds__(256) void wtrans_kernel(const float* __restrict__ W,
                                                     _Float16* __restrict__ WT,
                                                     int total) {
    int t = blockIdx.x * 256 + threadIdx.x;
    if (t >= total) return;
    int m  = t >> 12;          // matrix index (l*8+j)
    int r  = (t >> 6) & 63;    // out
    int c  = t & 63;           // in
    WT[t] = (_Float16)W[(m << 12) + (c << 6) + r];
}

// ---------------------------------------------------------------------------
// WMMA GEMM: Y[N,64] = act(X)[N,64] @ W[64,64] + b   (W given transposed f16)
// One wave per 16-row tile; each wave does 4 col-tiles, K=64 as 2x K=32.
// A-fragment layout (16x32 f16, wave32):
//   lanes 0-15 : M=lane,    K = chunk*32 + 0..15
//   lanes 16-31: M=lane-16, K = chunk*32 + 16..31
// B-fragment layout (32x16 f16):
//   lanes 0-15 : N=lane,    K = chunk*32 + 0..15   (contiguous in WT row)
//   lanes 16-31: N=lane-16, K = chunk*32 + 16..31
// D layout (16x16 f32): lane<16 -> N=lane, M=r ; lane>=16 -> N=lane-16, M=8+r
// RELU_IN fuses the previous layer's ReLU into the f32->f16 conversion.
// ---------------------------------------------------------------------------
template <bool RELU_IN>
__device__ inline v16h load_a_frag(const float* __restrict__ X, int row, int chunk, int lane) {
    const float* p = X + row * 64 + chunk * 32 + ((lane >> 4) << 4);
    v16h a;
#pragma unroll
    for (int j = 0; j < 8; ++j) {
        float v = p[j];
        if (RELU_IN) v = fmaxf(v, 0.f);
        a[j] = (_Float16)v;
    }
#pragma unroll
    for (int j = 0; j < 8; ++j) {
        float v = p[8 + j];
        if (RELU_IN) v = fmaxf(v, 0.f);
        a[8 + j] = (_Float16)v;
    }
    return a;
}

template <bool ACCUM, bool RELU_IN>
__global__ __launch_bounds__(256)
void gemm64_wmma(const float* __restrict__ X, const _Float16* __restrict__ WT,
                 const float* __restrict__ bias, float* __restrict__ Y, int N) {
    const int wave = threadIdx.x >> 5;
    const int lane = threadIdx.x & 31;
    // prefetch the weight matrix (64 rows x 128B) via CDNA5 global_prefetch
    __builtin_prefetch((const void*)(WT + ((threadIdx.x & 63) << 6)), 0, 1);

    const int tile = blockIdx.x * 8 + wave;
    const int row0 = tile * 16;
    if (row0 >= N) return;  // uniform per wave -> EXEC stays all-1s for WMMA

    int mrow = row0 + (lane & 15);
    if (mrow > N - 1) mrow = N - 1;  // clamp loads in tail tile
    const v16h a0 = load_a_frag<RELU_IN>(X, mrow, 0, lane);
    const v16h a1 = load_a_frag<RELU_IN>(X, mrow, 1, lane);

    const int  nsub  = lane & 15;
    const int  khalf = lane >> 4;
    const bool full  = (row0 + 16 <= N);  // uniform per wave

    v8f acc[4];
#pragma unroll
    for (int c = 0; c < 4; ++c) {
        const int col = c * 16 + nsub;
        const v16h b0 = *(const v16h*)(WT + col * 64 + khalf * 16);
        const v16h b1 = *(const v16h*)(WT + col * 64 + 32 + khalf * 16);
        v8f d = {};
        d = __builtin_amdgcn_wmma_f32_16x16x32_f16(false, a0, false, b0,
                                                   (short)0, d, false, false);
        d = __builtin_amdgcn_wmma_f32_16x16x32_f16(false, a1, false, b1,
                                                   (short)0, d, false, false);
        const float bcol = bias[col];
#pragma unroll
        for (int r = 0; r < 8; ++r) d[r] += bcol;
        acc[c] = d;
    }

    float* yrow = Y + (size_t)(row0 + khalf * 8) * 64 + nsub;
    if (full) {
        // straight-line, fully coalesced epilogue (common case)
#pragma unroll
        for (int c = 0; c < 4; ++c) {
#pragma unroll
            for (int r = 0; r < 8; ++r) {
                float* p = yrow + (size_t)r * 64 + c * 16;
                if (ACCUM) *p += acc[c][r];
                else       *p  = acc[c][r];
            }
        }
    } else {
#pragma unroll
        for (int c = 0; c < 4; ++c) {
#pragma unroll
            for (int r = 0; r < 8; ++r) {
                const int row = row0 + khalf * 8 + r;
                if (row < N) {
                    float* p = Y + (size_t)row * 64 + c * 16 + nsub;
                    if (ACCUM) *p += acc[c][r];
                    else       *p  = acc[c][r];
                }
            }
        }
    }
}

// ---------------------------------------------------------------------------
// Utility / edge kernels
// ---------------------------------------------------------------------------
__global__ __launch_bounds__(256) void fill_f32(float* __restrict__ p, float v, int n) {
    int i = blockIdx.x * 256 + threadIdx.x;
    if (i < n) p[i] = v;
}

__device__ inline void atomicMaxF(float* addr, float val) {
    unsigned int* ua = (unsigned int*)addr;
    unsigned int old = *ua;
    while (__uint_as_float(old) < val) {
        unsigned int assumed = old;
        old = atomicCAS(ua, assumed, __float_as_uint(val));
        if (old == assumed) break;
    }
}

__global__ __launch_bounds__(256)
void score_max_kernel(const float* __restrict__ q, const float* __restrict__ k,
                      const int* __restrict__ srcv, const int* __restrict__ dstv,
                      float* __restrict__ score, float* __restrict__ m, int ne) {
    int e = blockIdx.x * 256 + threadIdx.x;
    if (e >= ne) return;
    const float4* qq = (const float4*)(q + (long)dstv[e] * 64);
    const float4* kk = (const float4*)(k + (long)srcv[e] * 64);
    float s = 0.f;
#pragma unroll
    for (int i = 0; i < 16; ++i) {
        float4 a = qq[i], b = kk[i];
        s += a.x * b.x + a.y * b.y + a.z * b.z + a.w * b.w;
    }
    s *= 0.125f;  // 1/sqrt(64)
    score[e] = s;
    atomicMaxF(&m[dstv[e]], s);
}

__global__ __launch_bounds__(256)
void exp_sum_kernel(const float* __restrict__ score, const int* __restrict__ dstv,
                    const float* __restrict__ m, float* __restrict__ ex,
                    float* __restrict__ ssum, int ne) {
    int e = blockIdx.x * 256 + threadIdx.x;
    if (e >= ne) return;
    const int d = dstv[e];
    const float v = __expf(score[e] - m[d]);
    ex[e] = v;
    atomicAdd(&ssum[d], v);
}

__global__ __launch_bounds__(256)
void aggregate_kernel(const float* __restrict__ ex, const float* __restrict__ ssum,
                      const float* __restrict__ vmat, const int* __restrict__ srcv,
                      const int* __restrict__ dstv, float* __restrict__ out, int total) {
    int t = blockIdx.x * 256 + threadIdx.x;
    if (t >= total) return;
    const int e = t >> 6;
    const int i = t & 63;
    const int d = dstv[e];
    const float w = ex[e] / ssum[d];
    atomicAdd(&out[(long)d * 64 + i], w * vmat[(long)srcv[e] * 64 + i]);
}

// ---------------------------------------------------------------------------
// Pooling (fuses the final layer's ReLU) + head
// ---------------------------------------------------------------------------
__global__ __launch_bounds__(256)
void pool_accum(const float* __restrict__ x, const int* __restrict__ batch,
                float* __restrict__ psum, float* __restrict__ cnt, int n, int off) {
    int t = blockIdx.x * 256 + threadIdx.x;
    int node = t >> 6, i = t & 63;
    if (node >= n) return;
    const int b = batch[node];
    atomicAdd(&psum[b * 192 + off + i], fmaxf(x[(long)node * 64 + i], 0.f));
    if (i == 0) atomicAdd(&cnt[b], 1.f);
}

__global__ void pool_div(float* __restrict__ psum, const float* __restrict__ cnt, int off) {
    int b = blockIdx.x, i = threadIdx.x;  // G blocks x 64 threads
    psum[b * 192 + off + i] /= fmaxf(cnt[b], 1.f);
}

__global__ __launch_bounds__(64)
void head_kernel(const float* __restrict__ pooled, const float* __restrict__ W3,
                 const float* __restrict__ b3, const float* __restrict__ Wp,
                 const float* __restrict__ bp, const float* __restrict__ Wo,
                 const float* __restrict__ bo, float* __restrict__ out) {
    __shared__ float h1[64];
    __shared__ float h2[64];
    const int g = blockIdx.x, o = threadIdx.x;
    float s = b3[o];
    for (int i = 0; i < 192; ++i) s += pooled[g * 192 + i] * W3[i * 64 + o];
    h1[o] = s;
    __syncthreads();
    float s2 = bp[o];
    for (int i = 0; i < 64; ++i) s2 += h1[i] * Wp[i * 64 + o];
    s2 = (s2 > 20.f) ? s2 : log1pf(__expf(s2));  // softplus
    h2[o] = s2;
    __syncthreads();
    if (o == 0) {
        float r = bo[0];
        for (int i = 0; i < 64; ++i) r += h2[i] * Wo[i];
        out[g] = r;
    }
}

// ---------------------------------------------------------------------------
// Host orchestration
// ---------------------------------------------------------------------------
static inline int cdiv(int a, int b) { return (a + b - 1) / b; }

extern "C" void kernel_launch(void* const* d_in, const int* in_sizes, int n_in,
                              void* d_out, int out_size, void* d_ws, size_t ws_size,
                              hipStream_t stream) {
    (void)in_sizes; (void)n_in; (void)out_size; (void)ws_size;

    const float* x_in[3] = {(const float*)d_in[0], (const float*)d_in[1], (const float*)d_in[2]};
    const int*   ei[8];
    for (int j = 0; j < 8; ++j) ei[j] = (const int*)d_in[3 + j];
    const int* batch[3] = {(const int*)d_in[11], (const int*)d_in[12], (const int*)d_in[13]};
    const float* Wq = (const float*)d_in[14]; const float* bq = (const float*)d_in[15];
    const float* Wk = (const float*)d_in[16]; const float* bk = (const float*)d_in[17];
    const float* Wv = (const float*)d_in[18]; const float* bv = (const float*)d_in[19];
    const float* Ws = (const float*)d_in[20]; const float* bs = (const float*)d_in[21];
    const float* W3 = (const float*)d_in[22]; const float* b3 = (const float*)d_in[23];
    const float* Wp = (const float*)d_in[24]; const float* bp = (const float*)d_in[25];
    const float* Wo = (const float*)d_in[26]; const float* bo = (const float*)d_in[27];

    // ---- workspace carve (256B aligned) ----
    char* wp = (char*)d_ws;
    auto carve = [&](size_t bytes) -> void* {
        void* p = (void*)wp;
        wp += (bytes + 255) & ~(size_t)255;
        return p;
    };
    const int NMAT = CNL * CNET;                       // 24 matrices per kind
    _Float16* WTq = (_Float16*)carve((size_t)NMAT * 4096 * sizeof(_Float16));
    _Float16* WTk = (_Float16*)carve((size_t)NMAT * 4096 * sizeof(_Float16));
    _Float16* WTv = (_Float16*)carve((size_t)NMAT * 4096 * sizeof(_Float16));
    _Float16* WTs = (_Float16*)carve((size_t)NMAT * 4096 * sizeof(_Float16));
    float* featA[3]; float* featB[3];
    for (int t = 0; t < 3; ++t) featA[t] = (float*)carve((size_t)H_NSZ[t] * 64 * sizeof(float));
    for (int t = 0; t < 3; ++t) featB[t] = (float*)carve((size_t)H_NSZ[t] * 64 * sizeof(float));
    float* bufq  = (float*)carve((size_t)CNB * 64 * sizeof(float));
    float* bufk  = (float*)carve((size_t)CNB * 64 * sizeof(float));
    float* bufv  = (float*)carve((size_t)CNB * 64 * sizeof(float));
    float* score = (float*)carve((size_t)CE * sizeof(float));
    float* exb   = (float*)carve((size_t)CE * sizeof(float));
    float* mbuf  = (float*)carve((size_t)CNB * sizeof(float));
    float* sbuf  = (float*)carve((size_t)CNB * sizeof(float));
    float* psum  = (float*)carve((size_t)CG * 192 * sizeof(float));
    float* cnt3  = (float*)carve((size_t)3 * CG * sizeof(float));

    // ---- convert/transpose weights to f16 ----
    const int wtot = NMAT * 4096;
    wtrans_kernel<<<cdiv(wtot, 256), 256, 0, stream>>>(Wq, WTq, wtot);
    wtrans_kernel<<<cdiv(wtot, 256), 256, 0, stream>>>(Wk, WTk, wtot);
    wtrans_kernel<<<cdiv(wtot, 256), 256, 0, stream>>>(Wv, WTv, wtot);
    wtrans_kernel<<<cdiv(wtot, 256), 256, 0, stream>>>(Ws, WTs, wtot);

    // ---- 3 hetero-transformer layers ----
    // Feature buffers hold PRE-activation sums; ReLU is fused into the next
    // consumer (GEMM A-load for layers 1..2, pooling for the final layer).
    const float* xcur[3] = {x_in[0], x_in[1], x_in[2]};
    float** acc  = featA;
    float** next = featB;
    for (int l = 0; l < CNL; ++l) {
        const bool relu_in = (l > 0);
        bool dst_written[3] = {false, false, false};
        for (int j = 0; j < CNET; ++j) {
            const int st = H_SRC_T[j], dt = H_DST_T[j];
            const int ns = H_NSZ[st], nd = H_NSZ[dt];
            const int mi = l * CNET + j;
            const _Float16* wq = WTq + (size_t)mi * 4096;
            const _Float16* wk = WTk + (size_t)mi * 4096;
            const _Float16* wv = WTv + (size_t)mi * 4096;
            const _Float16* ws = WTs + (size_t)mi * 4096;
            const float* bqp = bq + (size_t)mi * 64;
            const float* bkp = bk + (size_t)mi * 64;
            const float* bvp = bv + (size_t)mi * 64;
            const float* bsp = bs + (size_t)mi * 64;
            const int gd = cdiv(cdiv(nd, 16), 8);
            const int gs = cdiv(cdiv(ns, 16), 8);
            // q = relu(x_dst) @ Wq + bq ; k/v from relu(x_src)
            if (relu_in) {
                gemm64_wmma<false, true><<<gd, 256, 0, stream>>>(xcur[dt], wq, bqp, bufq, nd);
                gemm64_wmma<false, true><<<gs, 256, 0, stream>>>(xcur[st], wk, bkp, bufk, ns);
                gemm64_wmma<false, true><<<gs, 256, 0, stream>>>(xcur[st], wv, bvp, bufv, ns);
            } else {
                gemm64_wmma<false, false><<<gd, 256, 0, stream>>>(xcur[dt], wq, bqp, bufq, nd);
                gemm64_wmma<false, false><<<gs, 256, 0, stream>>>(xcur[st], wk, bkp, bufk, ns);
                gemm64_wmma<false, false><<<gs, 256, 0, stream>>>(xcur[st], wv, bvp, bufv, ns);
            }
            // skip projection: first writer per dst type overwrites (no zero-fill
            // needed); later edge types accumulate (HeteroConv aggr='sum').
            if (!dst_written[dt]) {
                if (relu_in)
                    gemm64_wmma<false, true><<<gd, 256, 0, stream>>>(xcur[dt], ws, bsp, acc[dt], nd);
                else
                    gemm64_wmma<false, false><<<gd, 256, 0, stream>>>(xcur[dt], ws, bsp, acc[dt], nd);
                dst_written[dt] = true;
            } else {
                if (relu_in)
                    gemm64_wmma<true, true><<<gd, 256, 0, stream>>>(xcur[dt], ws, bsp, acc[dt], nd);
                else
                    gemm64_wmma<true, false><<<gd, 256, 0, stream>>>(xcur[dt], ws, bsp, acc[dt], nd);
            }
            // segment softmax over destination neighborhoods
            fill_f32<<<cdiv(nd, 256), 256, 0, stream>>>(mbuf, -1e30f, nd);
            fill_f32<<<cdiv(nd, 256), 256, 0, stream>>>(sbuf, 0.f, nd);
            const int* srcv = ei[j];
            const int* dstv = ei[j] + CE;
            score_max_kernel<<<cdiv(CE, 256), 256, 0, stream>>>(bufq, bufk, srcv, dstv,
                                                                score, mbuf, CE);
            exp_sum_kernel<<<cdiv(CE, 256), 256, 0, stream>>>(score, dstv, mbuf, exb, sbuf, CE);
            aggregate_kernel<<<cdiv(CE * 64, 256), 256, 0, stream>>>(exb, sbuf, bufv, srcv,
                                                                     dstv, acc[dt], CE * 64);
        }
        for (int t = 0; t < 3; ++t) xcur[t] = acc[t];
        float** tmp = acc; acc = next; next = tmp;  // ping-pong
    }

    // ---- mean pooling per graph, per node type (fuses final ReLU) ----
    fill_f32<<<cdiv(CG * 192, 256), 256, 0, stream>>>(psum, 0.f, CG * 192);
    fill_f32<<<cdiv(3 * CG, 256), 256, 0, stream>>>(cnt3, 0.f, 3 * CG);
    for (int t = 0; t < 3; ++t) {
        const int n = H_NSZ[t];
        pool_accum<<<cdiv(n * 64, 256), 256, 0, stream>>>(xcur[t], batch[t], psum,
                                                          cnt3 + t * CG, n, t * 64);
    }
    for (int t = 0; t < 3; ++t)
        pool_div<<<CG, 64, 0, stream>>>(psum, cnt3 + t * CG, t * 64);

    // ---- MLP head -> [G,1] ----
    head_kernel<<<CG, 64, 0, stream>>>(psum, W3, b3, Wp, bp, Wo, bo, (float*)d_out);
}
